// IntermediateSubgraphClassifier_26731876451139
// MI455X (gfx1250) — compile-verified, matrix operand order checked by reading, not run on hardware
//
#include <hip/hip_runtime.h>

// ---------------- problem constants ----------------
#define N_NODES   20000
#define N_EDGES   320000
#define EN        (N_EDGES + N_NODES)   // edges + self loops
#define DIM       1280
#define HID       64
#define HEADS     4
#define N_GR      256
#define NEG_SLOPE 0.2f

typedef __attribute__((ext_vector_type(16))) __bf16 v16bf;
typedef __attribute__((ext_vector_type(8)))  __bf16 v8bf;
typedef __attribute__((ext_vector_type(8)))  float  v8f;

// ---------------- workspace layout (bytes) ----------------
constexpr size_t AL = 256;
constexpr size_t alup(size_t x) { return (x + AL - 1) & ~(AL - 1); }

constexpr size_t OFF_XBF  = 0;                                       // x in bf16 [N,1280]
constexpr size_t OFF_W1T  = OFF_XBF  + alup((size_t)N_NODES*DIM*2);  // [Wl1|Wr1]^T bf16 [512,1280]
constexpr size_t OFF_B1C  = OFF_W1T  + alup((size_t)512*DIM*2);      // concat(bl1,br1) f32 [512]
constexpr size_t OFF_XLR1 = OFF_B1C  + alup((size_t)512*4);          // [N,512] f32 (xl|xr)
constexpr size_t OFF_LOG1 = OFF_XLR1 + alup((size_t)N_NODES*512*4);  // [EN,4] f32 logits->exp
constexpr size_t OFF_W2T  = OFF_LOG1 + alup((size_t)EN*4*4);         // [128,256] bf16
constexpr size_t OFF_B2C  = OFF_W2T  + alup((size_t)128*256*2);      // [128] f32
constexpr size_t OFF_XLR2 = OFF_B2C  + alup((size_t)128*4);          // [N,128] f32
constexpr size_t OFF_H1BF = OFF_XLR2 + alup((size_t)N_NODES*128*4);  // [N,256] bf16
constexpr size_t OFF_LOG2 = OFF_H1BF + alup((size_t)N_NODES*256*2);  // [EN] f32
// ---- zero-initialized (atomic-target) region, contiguous ----
constexpr size_t OFF_M1   = OFF_LOG2 + alup((size_t)EN*4);           // [N,4] u32 (encoded max)
constexpr size_t OFF_DEN1 = OFF_M1   + alup((size_t)N_NODES*4*4);    // [N,4] f32
constexpr size_t OFF_H1   = OFF_DEN1 + alup((size_t)N_NODES*4*4);    // [N,256] f32 aggregation
constexpr size_t OFF_M2   = OFF_H1   + alup((size_t)N_NODES*256*4);  // [N] u32
constexpr size_t OFF_DEN2 = OFF_M2   + alup((size_t)N_NODES*4);      // [N] f32
constexpr size_t OFF_H2   = OFF_DEN2 + alup((size_t)N_NODES*4);      // [N,64] f32
constexpr size_t OFF_G    = OFF_H2   + alup((size_t)N_NODES*64*4);   // [256,64] f32
constexpr size_t OFF_END  = OFF_G    + alup((size_t)N_GR*64*4);
constexpr size_t ZERO_FLOATS = (OFF_END - OFF_M1) / 4;

// ---------------- small helpers ----------------
__device__ __forceinline__ void edge_sd(const int* __restrict__ ei, int e, int& s, int& d) {
  if (e < N_EDGES) { s = ei[e]; d = ei[N_EDGES + e]; }
  else             { s = e - N_EDGES; d = s; }            // appended self loops
}
__device__ __forceinline__ unsigned enc_f(float f) {      // order-preserving float->uint
  unsigned u = __float_as_uint(f);
  return (u & 0x80000000u) ? ~u : (u | 0x80000000u);
}
__device__ __forceinline__ float dec_f(unsigned u) {
  return (u & 0x80000000u) ? __uint_as_float(u ^ 0x80000000u) : __uint_as_float(~u);
}
__device__ __forceinline__ float lrelu(float v) { return v > 0.f ? v : NEG_SLOPE * v; }

// ---------------- utility kernels ----------------
__global__ void k_zero(float* __restrict__ p, size_t n) {
  for (size_t i = (size_t)blockIdx.x * blockDim.x + threadIdx.x; i < n;
       i += (size_t)gridDim.x * blockDim.x) p[i] = 0.f;
}

// 8-wide f32 -> bf16 conversion (16B stores); n must be a multiple of 8
__global__ void k_f32_to_bf16_v8(const float4* __restrict__ src, v8bf* __restrict__ dst,
                                 size_t n8) {
  for (size_t i = (size_t)blockIdx.x * blockDim.x + threadIdx.x; i < n8;
       i += (size_t)gridDim.x * blockDim.x) {
    float4 f0 = src[2 * i], f1 = src[2 * i + 1];
    v8bf o;
    o[0] = (__bf16)f0.x; o[1] = (__bf16)f0.y; o[2] = (__bf16)f0.z; o[3] = (__bf16)f0.w;
    o[4] = (__bf16)f1.x; o[5] = (__bf16)f1.y; o[6] = (__bf16)f1.z; o[7] = (__bf16)f1.w;
    dst[i] = o;
  }
}

// wt[n*K+k] = (n<halfN ? Wl[k][n] : Wr[k][n-halfN]) ; W* row-major [K, halfN]
__global__ void k_build_wt(const float* __restrict__ Wl, const float* __restrict__ Wr,
                           __bf16* __restrict__ wt, int K, int halfN) {
  size_t total = (size_t)(2 * halfN) * K;
  for (size_t i = (size_t)blockIdx.x * blockDim.x + threadIdx.x; i < total;
       i += (size_t)gridDim.x * blockDim.x) {
    int n = (int)(i / K), k = (int)(i % K);
    float v = (n < halfN) ? Wl[(size_t)k * halfN + n] : Wr[(size_t)k * halfN + (n - halfN)];
    wt[i] = (__bf16)v;
  }
}

__global__ void k_bias_cat(const float* __restrict__ bl, const float* __restrict__ br,
                           float* __restrict__ out, int halfN) {
  int i = blockIdx.x * blockDim.x + threadIdx.x;
  if (i < 2 * halfN) out[i] = (i < halfN) ? bl[i] : br[i - halfN];
}

// ------- bf16 WMMA GEMM: C[M,NC] = A[M,K] * BT[NC,K]^T + bias --------------------
// One wave per 16-row x 128-col strip: the A fragment (big streaming matrix) is
// loaded once per k-step and reused across 8 WMMAs; BT (small, WGP$/L2-resident)
// is re-read from cache.  K is a template constant so the 8 B-tile displacements
// (t*16*K*2 bytes, < 8MB) fold into global_load_b128 immediate offsets, and all
// 16 B loads per k-step are issued as one batch so WMMAs overlap the load clause.
// A fragment (16-bit A 16x32, ISA 7.12.2): lane l -> row l&15, K = koff..koff+7 and
// koff+16..koff+23 with koff = 8*(l>>4).  BT stored row-per-output-column => same.
template <int K>
__global__ void __launch_bounds__(256)
k_gemm_wmma(const __bf16* __restrict__ A, const __bf16* __restrict__ BT,
            const float* __restrict__ bias, float* __restrict__ C,
            int M, int NC, int cgshift) {
  int wave = (blockIdx.x * blockDim.x + threadIdx.x) >> 5;
  int lane = threadIdx.x & 31;
  int tm = wave >> cgshift;
  int cg = wave & ((1 << cgshift) - 1);
  if (tm * 16 >= M) return;

  int half8 = (lane >> 4) << 3;                     // 0 or 8
  const __bf16* pA = A  + (size_t)(tm * 16 + (lane & 15)) * K + half8;
  const __bf16* pB = BT + (size_t)(cg * 128 + (lane & 15)) * K + half8;

  v8f acc[8] = {};
  for (int k = 0; k < K; k += 32) {
    v8bf a0 = *(const v8bf*)(pA + k);
    v8bf a1 = *(const v8bf*)(pA + k + 16);
    v16bf a = __builtin_shufflevector(a0, a1, 0,1,2,3,4,5,6,7,8,9,10,11,12,13,14,15);
    v16bf bfrag[8];
#pragma unroll
    for (int t = 0; t < 8; ++t) {                   // batch all B loads (one clause)
      v8bf b0 = *(const v8bf*)(pB + k + t * 16 * K);
      v8bf b1 = *(const v8bf*)(pB + k + 16 + t * 16 * K);
      bfrag[t] = __builtin_shufflevector(b0, b1, 0,1,2,3,4,5,6,7,8,9,10,11,12,13,14,15);
    }
#pragma unroll
    for (int t = 0; t < 8; ++t)
      acc[t] = __builtin_amdgcn_wmma_f32_16x16x32_bf16(false, a, false, bfrag[t],
                                                       (short)0, acc[t], false, false);
  }

  int rbase = tm * 16 + half8;                      // C layout: VGPR r -> row r (+8 hi half)
#pragma unroll
  for (int t = 0; t < 8; ++t) {
    int col = (cg * 8 + t) * 16 + (lane & 15);
    float bv = bias[col];
#pragma unroll
    for (int r = 0; r < 8; ++r)
      C[(size_t)(rbase + r) * NC + col] = acc[t][r] + bv;
  }
}

// ---------------- layer-1 attention (H=4, C=64), one wave per edge ----------------
__global__ void __launch_bounds__(256)
k_logits1(const int* __restrict__ ei, const float* __restrict__ xlr,
          const float* __restrict__ att, float* __restrict__ logit,
          unsigned* __restrict__ mmax) {
  int e = (blockIdx.x * blockDim.x + threadIdx.x) >> 5;
  int lane = threadIdx.x & 31;
  if (e >= EN) return;
  int s, d; edge_sd(ei, e, s, d);
  const float* ps = xlr + (size_t)s * 512;          // Wl x_j  (cols 0..255)
  const float* pd = xlr + (size_t)d * 512 + 256;    // Wr x_i  (cols 256..511)
  float acc[HEADS] = {0.f, 0.f, 0.f, 0.f};
#pragma unroll
  for (int j = 0; j < 8; ++j) {                     // head = j>>1, uniform per iteration
    int idx = j * 32 + lane;
    float v = lrelu(ps[idx] + pd[idx]);
    acc[j >> 1] += v * att[(j >> 1) * 64 + (idx & 63)];
  }
#pragma unroll
  for (int h = 0; h < HEADS; ++h) {
    float a = acc[h];
#pragma unroll
    for (int off = 16; off; off >>= 1) a += __shfl_xor(a, off, 32);
    if (lane == 0) {
      logit[(size_t)e * 4 + h] = a;
      atomicMax(&mmax[(size_t)d * 4 + h], enc_f(a));
    }
  }
}

__global__ void k_exp1(const int* __restrict__ ei, float* __restrict__ logit,
                       const unsigned* __restrict__ mmax, float* __restrict__ denom) {
  size_t total = (size_t)EN * 4;
  for (size_t t = (size_t)blockIdx.x * blockDim.x + threadIdx.x; t < total;
       t += (size_t)gridDim.x * blockDim.x) {
    int e = (int)(t >> 2), h = (int)(t & 3);
    int s, d; edge_sd(ei, e, s, d); (void)s;
    float ex = expf(logit[t] - dec_f(mmax[(size_t)d * 4 + h]));
    logit[t] = ex;
    atomicAdd(&denom[(size_t)d * 4 + h], ex);
  }
}

__global__ void __launch_bounds__(256)
k_agg1(const int* __restrict__ ei, const float* __restrict__ xlr,
       const float* __restrict__ exv, const float* __restrict__ denom,
       float* __restrict__ out) {
  int e = (blockIdx.x * blockDim.x + threadIdx.x) >> 5;
  int lane = threadIdx.x & 31;
  if (e >= EN) return;
  int s, d; edge_sd(ei, e, s, d);
  float alpha[HEADS];
#pragma unroll
  for (int h = 0; h < HEADS; ++h)
    alpha[h] = exv[(size_t)e * 4 + h] / (denom[(size_t)d * 4 + h] + 1e-16f);
  const float* ps = xlr + (size_t)s * 512;
  float* po = out + (size_t)d * 256;
#pragma unroll
  for (int j = 0; j < 8; ++j) {
    int idx = j * 32 + lane;
    atomicAdd(&po[idx], alpha[j >> 1] * ps[idx]);
  }
}

// h1bf = bf16(relu(agg + bias1)), 8-wide; n8 = n/8, channel mask on 8-element base
__global__ void k_relu_bias_bf16_v8(const float* __restrict__ acc,
                                    const float* __restrict__ bias,
                                    v8bf* __restrict__ dst, size_t n8, int cmask) {
  for (size_t i = (size_t)blockIdx.x * blockDim.x + threadIdx.x; i < n8;
       i += (size_t)gridDim.x * blockDim.x) {
    const float4* p = (const float4*)(acc + i * 8);
    float4 f0 = p[0], f1 = p[1];
    int c = (int)((i * 8) & (size_t)cmask);
    float vv[8] = {f0.x, f0.y, f0.z, f0.w, f1.x, f1.y, f1.z, f1.w};
    v8bf o;
#pragma unroll
    for (int j = 0; j < 8; ++j) {
      float v = vv[j] + bias[c + j];
      o[j] = (__bf16)(v > 0.f ? v : 0.f);
    }
    dst[i] = o;
  }
}

// ---------------- layer-2 attention (H=1, C=64) ----------------
__global__ void __launch_bounds__(256)
k_logits2(const int* __restrict__ ei, const float* __restrict__ xlr,
          const float* __restrict__ att, float* __restrict__ logit,
          unsigned* __restrict__ mmax) {
  int e = (blockIdx.x * blockDim.x + threadIdx.x) >> 5;
  int lane = threadIdx.x & 31;
  if (e >= EN) return;
  int s, d; edge_sd(ei, e, s, d);
  const float* ps = xlr + (size_t)s * 128;
  const float* pd = xlr + (size_t)d * 128 + 64;
  float a = 0.f;
#pragma unroll
  for (int j = 0; j < 2; ++j) {
    int idx = j * 32 + lane;
    a += lrelu(ps[idx] + pd[idx]) * att[idx];
  }
#pragma unroll
  for (int off = 16; off; off >>= 1) a += __shfl_xor(a, off, 32);
  if (lane == 0) {
    logit[e] = a;
    atomicMax(&mmax[d], enc_f(a));
  }
}

__global__ void k_exp2(const int* __restrict__ ei, float* __restrict__ logit,
                       const unsigned* __restrict__ mmax, float* __restrict__ denom) {
  for (size_t t = (size_t)blockIdx.x * blockDim.x + threadIdx.x; t < (size_t)EN;
       t += (size_t)gridDim.x * blockDim.x) {
    int e = (int)t;
    int s, d; edge_sd(ei, e, s, d); (void)s;
    float ex = expf(logit[t] - dec_f(mmax[d]));
    logit[t] = ex;
    atomicAdd(&denom[d], ex);
  }
}

__global__ void __launch_bounds__(256)
k_agg2(const int* __restrict__ ei, const float* __restrict__ xlr,
       const float* __restrict__ exv, const float* __restrict__ denom,
       float* __restrict__ out) {
  int e = (blockIdx.x * blockDim.x + threadIdx.x) >> 5;
  int lane = threadIdx.x & 31;
  if (e >= EN) return;
  int s, d; edge_sd(ei, e, s, d);
  float alpha = exv[e] / (denom[d] + 1e-16f);
  const float* ps = xlr + (size_t)s * 128;
  float* po = out + (size_t)d * 64;
#pragma unroll
  for (int j = 0; j < 2; ++j) {
    int idx = j * 32 + lane;
    atomicAdd(&po[idx], alpha * ps[idx]);
  }
}

// global_add_pool of relu(h2 + bias2)
__global__ void k_pool(const float* __restrict__ h2, const float* __restrict__ bias,
                       const int* __restrict__ batch, float* __restrict__ g) {
  size_t total = (size_t)N_NODES * 64;
  for (size_t i = (size_t)blockIdx.x * blockDim.x + threadIdx.x; i < total;
       i += (size_t)gridDim.x * blockDim.x) {
    int row = (int)(i >> 6), c = (int)(i & 63);
    float v = h2[i] + bias[c];
    v = v > 0.f ? v : 0.f;
    atomicAdd(&g[(size_t)batch[row] * 64 + c], v);
  }
}

// MLP head: out[b] = relu(g_b @ W3 + b3) @ W4 + b4 ; one 64-thread block per graph
__global__ void __launch_bounds__(64)
k_head(const float* __restrict__ g, const float* __restrict__ W3, const float* __restrict__ b3,
       const float* __restrict__ W4, const float* __restrict__ b4, float* __restrict__ out) {
  __shared__ float sred[64];
  int t = threadIdx.x, blk = blockIdx.x;
  const float* gr = g + (size_t)blk * 64;
  float hid = b3[t];
#pragma unroll
  for (int c = 0; c < 64; ++c) hid += gr[c] * W3[c * 64 + t];
  hid = hid > 0.f ? hid : 0.f;
  sred[t] = hid * W4[t];
  __syncthreads();
  if (t < 32) {
    float v = sred[t] + sred[t + 32];
#pragma unroll
    for (int off = 16; off; off >>= 1) v += __shfl_xor(v, off, 32);
    if (t == 0) out[blk] = v + b4[0];
  }
}

// ---------------- launcher ----------------
extern "C" void kernel_launch(void* const* d_in, const int* in_sizes, int n_in,
                              void* d_out, int out_size, void* d_ws, size_t ws_size,
                              hipStream_t stream) {
  (void)in_sizes; (void)n_in; (void)out_size; (void)ws_size;
  const float* x     = (const float*)d_in[0];
  const int*   ei    = (const int*)  d_in[1];
  const int*   batch = (const int*)  d_in[2];
  const float* Wl1   = (const float*)d_in[3];
  const float* bl1   = (const float*)d_in[4];
  const float* Wr1   = (const float*)d_in[5];
  const float* br1   = (const float*)d_in[6];
  const float* att1  = (const float*)d_in[7];
  const float* bias1 = (const float*)d_in[8];
  const float* Wl2   = (const float*)d_in[9];
  const float* bl2   = (const float*)d_in[10];
  const float* Wr2   = (const float*)d_in[11];
  const float* br2   = (const float*)d_in[12];
  const float* att2  = (const float*)d_in[13];
  const float* bias2 = (const float*)d_in[14];
  const float* W3    = (const float*)d_in[15];
  const float* b3    = (const float*)d_in[16];
  const float* W4    = (const float*)d_in[17];
  const float* b4    = (const float*)d_in[18];
  float* out = (float*)d_out;

  char* ws = (char*)d_ws;
  __bf16*   xbf   = (__bf16*)  (ws + OFF_XBF);
  __bf16*   w1t   = (__bf16*)  (ws + OFF_W1T);
  float*    b1c   = (float*)   (ws + OFF_B1C);
  float*    xlr1  = (float*)   (ws + OFF_XLR1);
  float*    log1  = (float*)   (ws + OFF_LOG1);
  __bf16*   w2t   = (__bf16*)  (ws + OFF_W2T);
  float*    b2c   = (float*)   (ws + OFF_B2C);
  float*    xlr2  = (float*)   (ws + OFF_XLR2);
  __bf16*   h1bf  = (__bf16*)  (ws + OFF_H1BF);
  float*    log2  = (float*)   (ws + OFF_LOG2);
  unsigned* m1    = (unsigned*)(ws + OFF_M1);
  float*    den1  = (float*)   (ws + OFF_DEN1);
  float*    h1    = (float*)   (ws + OFF_H1);
  unsigned* m2    = (unsigned*)(ws + OFF_M2);
  float*    den2  = (float*)   (ws + OFF_DEN2);
  float*    h2    = (float*)   (ws + OFF_H2);
  float*    g     = (float*)   (ws + OFF_G);

  const int B = 256;
  const int EDGE_BLOCKS = (EN + 7) / 8;                 // 8 waves/block, 1 wave/edge

  // zero all atomic-target buffers (contiguous region) every call
  k_zero<<<2048, B, 0, stream>>>((float*)(ws + OFF_M1), ZERO_FLOATS);

  // precision conversion + weight transpose/concat
  k_f32_to_bf16_v8<<<2048, B, 0, stream>>>((const float4*)x, (v8bf*)xbf,
                                           (size_t)N_NODES * DIM / 8);
  k_build_wt<<<1280, B, 0, stream>>>(Wl1, Wr1, w1t, DIM, HEADS * HID);
  k_bias_cat<<<2, B, 0, stream>>>(bl1, br1, b1c, HEADS * HID);
  k_build_wt<<<128, B, 0, stream>>>(Wl2, Wr2, w2t, HEADS * HID, HID);
  k_bias_cat<<<1, 128, 0, stream>>>(bl2, br2, b2c, HID);

  // layer 1: GEMM waves = (M/16) * (NC/128) = 1250 * 4
  k_gemm_wmma<DIM><<<(1250 * 4 + 7) / 8, B, 0, stream>>>(xbf, w1t, b1c, xlr1,
                                                         N_NODES, 512, 2);
  k_logits1<<<EDGE_BLOCKS, B, 0, stream>>>(ei, xlr1, att1, log1, m1);
  k_exp1<<<2048, B, 0, stream>>>(ei, log1, m1, den1);
  k_agg1<<<EDGE_BLOCKS, B, 0, stream>>>(ei, xlr1, log1, den1, h1);
  k_relu_bias_bf16_v8<<<2048, B, 0, stream>>>(h1, bias1, (v8bf*)h1bf,
                                              (size_t)N_NODES * 256 / 8, 255);

  // layer 2: GEMM waves = 1250 * 1
  k_gemm_wmma<HEADS * HID><<<(1250 + 7) / 8, B, 0, stream>>>(h1bf, w2t, b2c, xlr2,
                                                             N_NODES, 128, 0);
  k_logits2<<<EDGE_BLOCKS, B, 0, stream>>>(ei, xlr2, att2, log2, m2);
  k_exp2<<<1024, B, 0, stream>>>(ei, log2, m2, den2);
  k_agg2<<<EDGE_BLOCKS, B, 0, stream>>>(ei, xlr2, log2, den2, h2);

  // pooling + MLP head
  k_pool<<<2048, B, 0, stream>>>(h2, bias2, batch, g);
  k_head<<<N_GR, 64, 0, stream>>>(g, W3, b3, W4, b4, out);
}
